// RGATSQL_34746285425224
// MI455X (gfx1250) — matrix operand order repaired and checked by reading, not compiled, not run on hardware
//
#include <hip/hip_runtime.h>

#define NN   50000
#define EE   800000
#define ND   128
#define HH   8
#define DKD  16
#define FFD  512
#define LLL  2

typedef __attribute__((ext_vector_type(16))) __bf16 v16bf;
typedef __attribute__((ext_vector_type(8)))  float  v8f;
typedef __attribute__((ext_vector_type(4)))  float  f32x4;
typedef __attribute__((ext_vector_type(4)))  unsigned int u32x4;
typedef __attribute__((ext_vector_type(2)))  unsigned int u32x2;

__device__ __forceinline__ unsigned short f2bf(float f) {
    unsigned int u = __float_as_uint(f);
    u += 0x7FFFu + ((u >> 16) & 1u);   // round-to-nearest-even
    return (unsigned short)(u >> 16);
}

union TileU { u32x4 u[2]; v16bf v; };

// A fragment (16x32 bf16, M = lane&15): lanes 0-15 hold K = k0+[0..7] then k0+16+[0..7];
// lanes 16-31 hold K = k0+8+[0..7] then k0+24+[0..7]  (per ISA 16-bit A layout)
__device__ __forceinline__ v16bf load_a_tile(const unsigned short* rowBase, int k0, int lane) {
    int off = (lane & 16) ? 8 : 0;
    TileU t;
    t.u[0] = *reinterpret_cast<const u32x4*>(rowBase + k0 + off);
    t.u[1] = *reinterpret_cast<const u32x4*>(rowBase + k0 + 16 + off);
    return t.v;
}

// B fragment (32x16 bf16, N = lane&15): lanes 0-15 hold K = k0+[0..7] then k0+16+[0..7]; lanes 16-31 K = k0+8+[0..7] then k0+24+[0..7]
__device__ __forceinline__ v16bf load_b_tile(const unsigned short* rowBase, int k0, int lane) {
    int off = (lane & 16) ? 8 : 0;
    TileU t;
    t.u[0] = *reinterpret_cast<const u32x4*>(rowBase + k0 + off);
    t.u[1] = *reinterpret_cast<const u32x4*>(rowBase + k0 + 16 + off);
    return t.v;
}

// y[M,Ncols] = A[M,K] @ W[Ncols,K]^T  (+bias) (+relu), outputs f32 and/or bf16.
// grid = (M/16, Ncols/128), block = 256 (8 waves, one 16x16 tile per wave)
__global__ __launch_bounds__(256) void rgat_gemm_nt_bf16(
        const unsigned short* __restrict__ A, const unsigned short* __restrict__ W,
        const float* __restrict__ bias, float* __restrict__ outF,
        unsigned short* __restrict__ outB, int M, int Ncols, int K, int doRelu) {
    int wave = threadIdx.x >> 5;
    int lane = threadIdx.x & 31;
    int row0 = blockIdx.x * 16;
    int col0 = (blockIdx.y * 8 + wave) * 16;
    if (row0 >= M || col0 >= Ncols) return;

    const unsigned short* aBase = A + (size_t)(row0 + (lane & 15)) * K;
    const unsigned short* bBase = W + (size_t)(col0 + (lane & 15)) * K;

    v8f acc = {};
    for (int k0 = 0; k0 < K; k0 += 32) {
        v16bf av = load_a_tile(aBase, k0, lane);
        v16bf bv = load_b_tile(bBase, k0, lane);
        acc = __builtin_amdgcn_wmma_f32_16x16x32_bf16(
                false, av, false, bv, (short)0, acc, false, false);
    }

    int   col   = col0 + (lane & 15);
    float biasv = bias ? bias[col] : 0.0f;
    int   rbase = row0 + ((lane >> 4) << 3);   // C/D layout: M = r + 8*(lane>=16)
#pragma unroll
    for (int r = 0; r < 8; ++r) {
        float v = acc[r] + biasv;
        if (doRelu) v = fmaxf(v, 0.0f);
        size_t idx = (size_t)(rbase + r) * Ncols + col;
        if (outF) outF[idx] = v;
        if (outB) outB[idx] = f2bf(v);
    }
}

// One wave32 per edge: 4 dims/lane, head = lane/4; shuffle-reduce dot within head,
// exp(clip(score/4)), scatter (v+e)*score and score with f32 atomics.
__global__ __launch_bounds__(256) void rgat_edge_attn(
        const float* __restrict__ q, const float* __restrict__ k, const float* __restrict__ v,
        const float* __restrict__ rel, const int* __restrict__ edges,
        const int* __restrict__ src, const int* __restrict__ dst,
        float* __restrict__ wv, float* __restrict__ z) {
    int wid  = (blockIdx.x * blockDim.x + threadIdx.x) >> 5;
    int lane = threadIdx.x & 31;
    if (wid >= EE) return;
    int s = src[wid], d = dst[wid], r = edges[wid];

    f32x4 kk = *reinterpret_cast<const f32x4*>(k + (size_t)s * ND + lane * 4);
    f32x4 qq = *reinterpret_cast<const f32x4*>(q + (size_t)d * ND + lane * 4);
    f32x4 ev = *reinterpret_cast<const f32x4*>(rel + (size_t)r * DKD + (lane & 3) * 4);

    float p = (kk.x + ev.x) * qq.x + (kk.y + ev.y) * qq.y +
              (kk.z + ev.z) * qq.z + (kk.w + ev.w) * qq.w;
    p += __shfl_xor(p, 1);
    p += __shfl_xor(p, 2);                       // sum over the 4 lanes of this head
    float score = __expf(fminf(fmaxf(p * 0.25f, -5.0f), 5.0f));  // /sqrt(16)

    f32x4 vv = *reinterpret_cast<const f32x4*>(v + (size_t)s * ND + lane * 4);
    float* wdst = wv + (size_t)d * ND + lane * 4;
    atomicAdd(wdst + 0, (vv.x + ev.x) * score);
    atomicAdd(wdst + 1, (vv.y + ev.y) * score);
    atomicAdd(wdst + 2, (vv.z + ev.z) * score);
    atomicAdd(wdst + 3, (vv.w + ev.w) * score);
    if ((lane & 3) == 0) atomicAdd(z + (size_t)d * HH + (lane >> 2), score);
}

// obf = bf16(wv / z[node,head])
__global__ void rgat_onorm_bf(const float* __restrict__ wv, const float* __restrict__ z,
                              unsigned short* __restrict__ obf) {
    int idx = blockIdx.x * blockDim.x + threadIdx.x;
    if (idx >= NN * ND) return;
    int node = idx >> 7;
    int head = (idx & 127) >> 4;
    obf[idx] = f2bf(wv[idx] / z[node * HH + head]);
}

// out = LayerNorm(x + t + bias) * g + b ; writes f32 and bf16. One wave per 128-wide row.
__global__ __launch_bounds__(256) void rgat_resid_ln(
        const float* __restrict__ x, const float* __restrict__ t, const float* __restrict__ bias,
        const float* __restrict__ g, const float* __restrict__ b,
        float* __restrict__ outF, unsigned short* __restrict__ outB) {
    int wave = threadIdx.x >> 5, lane = threadIdx.x & 31;
    int row = blockIdx.x * 8 + wave;
    if (row >= NN) return;

    f32x4 xv = *reinterpret_cast<const f32x4*>(x + (size_t)row * ND + lane * 4);
    f32x4 tv = *reinterpret_cast<const f32x4*>(t + (size_t)row * ND + lane * 4);
    f32x4 bv = *reinterpret_cast<const f32x4*>(bias + lane * 4);
    float u0 = xv.x + tv.x + bv.x, u1 = xv.y + tv.y + bv.y;
    float u2 = xv.z + tv.z + bv.z, u3 = xv.w + tv.w + bv.w;

    float s = u0 + u1 + u2 + u3;
#pragma unroll
    for (int m = 1; m < 32; m <<= 1) s += __shfl_xor(s, m);
    float mean = s * (1.0f / 128.0f);
    float d0 = u0 - mean, d1 = u1 - mean, d2 = u2 - mean, d3 = u3 - mean;
    float qv = d0 * d0 + d1 * d1 + d2 * d2 + d3 * d3;
#pragma unroll
    for (int m = 1; m < 32; m <<= 1) qv += __shfl_xor(qv, m);
    float inv = rsqrtf(qv * (1.0f / 128.0f) + 1e-5f);

    f32x4 gv = *reinterpret_cast<const f32x4*>(g + lane * 4);
    f32x4 bb = *reinterpret_cast<const f32x4*>(b + lane * 4);
    float y0 = d0 * inv * gv.x + bb.x, y1 = d1 * inv * gv.y + bb.y;
    float y2 = d2 * inv * gv.z + bb.z, y3 = d3 * inv * gv.w + bb.w;

    f32x4 yo; yo.x = y0; yo.y = y1; yo.z = y2; yo.w = y3;
    *reinterpret_cast<f32x4*>(outF + (size_t)row * ND + lane * 4) = yo;
    u32x2 pk;
    pk.x = (unsigned)f2bf(y0) | ((unsigned)f2bf(y1) << 16);
    pk.y = (unsigned)f2bf(y2) | ((unsigned)f2bf(y3) << 16);
    *reinterpret_cast<u32x2*>(outB + (size_t)row * ND + lane * 4) = pk;
}

__global__ void rgat_f32_to_bf16(const float* __restrict__ in, unsigned short* __restrict__ out, int n) {
    int i = blockIdx.x * blockDim.x + threadIdx.x;
    if (i < n) out[i] = f2bf(in[i]);
}

extern "C" void kernel_launch(void* const* d_in, const int* in_sizes, int n_in,
                              void* d_out, int out_size, void* d_ws, size_t ws_size,
                              hipStream_t stream) {
    const float* x_in  = (const float*)d_in[0];
    const int*   edges = (const int*)d_in[1];
    const int*   src   = (const int*)d_in[2];
    const int*   dst   = (const int*)d_in[3];
    const float* rel   = (const float*)d_in[4];
    const float* Wq    = (const float*)d_in[5];
    const float* bq    = (const float*)d_in[6];
    const float* Wk    = (const float*)d_in[7];
    const float* Wv    = (const float*)d_in[8];
    const float* Wo    = (const float*)d_in[9];
    const float* bo    = (const float*)d_in[10];
    const float* ln1g  = (const float*)d_in[11];
    const float* ln1b  = (const float*)d_in[12];
    const float* W1    = (const float*)d_in[13];
    const float* b1    = (const float*)d_in[14];
    const float* W2    = (const float*)d_in[15];
    const float* b2    = (const float*)d_in[16];
    const float* ln2g  = (const float*)d_in[17];
    const float* ln2b  = (const float*)d_in[18];

    const size_t szF = (size_t)NN * ND * sizeof(float);       // 25.6 MB
    const size_t szH = (size_t)NN * ND * sizeof(unsigned short);

    char* p = (char*)d_ws;
    float* xbuf  = (float*)p; p += szF;
    float* qb    = (float*)p; p += szF;
    float* kb    = (float*)p; p += szF;
    float* vb    = (float*)p; p += szF;
    float* wvb   = (float*)p; p += szF;
    float* tmpb  = (float*)p; p += szF;
    float* outln = (float*)p; p += szF;
    float* zb    = (float*)p; p += (size_t)NN * HH * sizeof(float);
    unsigned short* xbf   = (unsigned short*)p; p += szH;
    unsigned short* obf   = (unsigned short*)p; p += szH;
    unsigned short* outbf = (unsigned short*)p; p += szH;
    unsigned short* hbf   = (unsigned short*)p; p += (size_t)NN * FFD * sizeof(unsigned short);
    unsigned short* wbf   = (unsigned short*)p;   // 2 layers * 196608 halves

    dim3 blk(256);
    // Convert all weights to bf16 (row-major [out, in] — B-fragment reads rows contiguously)
    for (int l = 0; l < LLL; ++l) {
        unsigned short* wl = wbf + (size_t)l * 196608;
        rgat_f32_to_bf16<<<(16384 + 255) / 256, blk, 0, stream>>>(Wq + (size_t)l * 16384, wl,          16384);
        rgat_f32_to_bf16<<<(16384 + 255) / 256, blk, 0, stream>>>(Wk + (size_t)l * 16384, wl + 16384,  16384);
        rgat_f32_to_bf16<<<(16384 + 255) / 256, blk, 0, stream>>>(Wv + (size_t)l * 16384, wl + 32768,  16384);
        rgat_f32_to_bf16<<<(16384 + 255) / 256, blk, 0, stream>>>(Wo + (size_t)l * 16384, wl + 49152,  16384);
        rgat_f32_to_bf16<<<(65536 + 255) / 256, blk, 0, stream>>>(W1 + (size_t)l * 65536, wl + 65536,  65536);
        rgat_f32_to_bf16<<<(65536 + 255) / 256, blk, 0, stream>>>(W2 + (size_t)l * 65536, wl + 131072, 65536);
    }
    rgat_f32_to_bf16<<<(NN * ND + 255) / 256, blk, 0, stream>>>(x_in, xbf, NN * ND);

    const float* xf = x_in;
    for (int l = 0; l < LLL; ++l) {
        unsigned short* wl = wbf + (size_t)l * 196608;
        dim3 g128(NN / 16, 1), g512(NN / 16, 4);

        // q = x@Wq.T + bq ; k = x@Wk.T ; v = x@Wv.T
        rgat_gemm_nt_bf16<<<g128, blk, 0, stream>>>(xbf, wl,         bq + l * ND, qb, nullptr, NN, ND, ND, 0);
        rgat_gemm_nt_bf16<<<g128, blk, 0, stream>>>(xbf, wl + 16384, nullptr,     kb, nullptr, NN, ND, ND, 0);
        rgat_gemm_nt_bf16<<<g128, blk, 0, stream>>>(xbf, wl + 32768, nullptr,     vb, nullptr, NN, ND, ND, 0);

        hipMemsetAsync(wvb, 0, szF, stream);
        hipMemsetAsync(zb,  0, (size_t)NN * HH * sizeof(float), stream);
        rgat_edge_attn<<<EE / 8, blk, 0, stream>>>(qb, kb, vb, rel, edges, src, dst, wvb, zb);
        rgat_onorm_bf<<<(NN * ND + 255) / 256, blk, 0, stream>>>(wvb, zb, obf);

        // tmp = o @ Wo.T ; out = LN(x + tmp + bo)
        rgat_gemm_nt_bf16<<<g128, blk, 0, stream>>>(obf, wl + 49152, nullptr, tmpb, nullptr, NN, ND, ND, 0);
        rgat_resid_ln<<<NN / 8, blk, 0, stream>>>(xf, tmpb, bo + l * ND,
                                                  ln1g + l * ND, ln1b + l * ND, outln, outbf);

        // h = relu(out @ W1.T + b1) -> bf16 ; tmp = h @ W2.T ; x' = LN(out + tmp + b2)
        rgat_gemm_nt_bf16<<<g512, blk, 0, stream>>>(outbf, wl + 65536, b1 + (size_t)l * FFD,
                                                    nullptr, hbf, NN, FFD, ND, 1);
        rgat_gemm_nt_bf16<<<g128, blk, 0, stream>>>(hbf, wl + 131072, nullptr, tmpb, nullptr, NN, ND, FFD, 0);

        float* xout = (l == LLL - 1) ? (float*)d_out : xbuf;
        rgat_resid_ln<<<NN / 8, blk, 0, stream>>>(outln, tmpb, b2 + l * ND,
                                                  ln2g + l * ND, ln2b + l * ND, xout, xbf);
        xf = xbuf;
    }
}
